// Block_3599182594491
// MI455X (gfx1250) — compile-verified, matrix operand order checked by reading, not run on hardware
//
#include <hip/hip_runtime.h>
#include <stdint.h>

typedef __attribute__((ext_vector_type(16))) __bf16 v16bf;
typedef __attribute__((ext_vector_type(8)))  float  v8f;
typedef __attribute__((ext_vector_type(4)))  unsigned v4u;
typedef __attribute__((ext_vector_type(8)))  unsigned v8u;

#define BB 4
#define TT 2048
#define CC 1024
#define HH 16
#define HS 64
#define MMROWS (BB*TT)     // 8192
#define FFDIM (4*CC)       // 4096

static __device__ __forceinline__ unsigned short f2bf(float f) {
    union { float f; unsigned u; } a; a.f = f;
    unsigned r = a.u + 0x7FFFu + ((a.u >> 16) & 1u);   // round-to-nearest-even
    return (unsigned short)(r >> 16);
}

union AF { uint4 q[2]; v16bf v; unsigned short u[16]; };

// Load one bf16 WMMA operand fragment from an LDS tile (16-bit operand layout,
// CDNA5 ISA: lanes 0-15 hold K {0..7,16..23}, lanes 16-31 hold K {8..15,24..31}).
template <int S>
static __device__ __forceinline__ v16bf ldfrag(const unsigned short* t, int row, int c0) {
    AF f;
    f.q[0] = *(const uint4*)(t + row * S + c0);
    f.q[1] = *(const uint4*)(t + row * S + c0 + 16);
    return f.v;
}

static __device__ __forceinline__ v8f wmma_bf16(v16bf a, v16bf b, v8f c) {
    return __builtin_amdgcn_wmma_f32_16x16x32_bf16(false, a, false, b, (short)0, c, false, false);
}

// CDNA5 async global->LDS copy (ASYNCcnt) — used by attention tile streaming.
static __device__ __forceinline__ void async_b128(const unsigned short* lds_ptr,
                                                  const unsigned short* gptr) {
    unsigned lds = (unsigned)(size_t)lds_ptr;              // low 32 bits = LDS offset
    unsigned long long g = (unsigned long long)(size_t)gptr;
    asm volatile("global_load_async_to_lds_b128 %0, %1, off"
                 :: "v"(lds), "v"(g) : "memory");
}
static __device__ __forceinline__ void wait_async(int n) {
    if (n == 0) asm volatile("s_wait_asynccnt 0x0" ::: "memory");
    else asm volatile("s_wait_asynccnt 0x8" ::: "memory");
}

// CDNA5 Tensor Data Mover: DMA one 2D bf16 tile (32 x 16 rows, row stride
// `stride` elements) from global to LDS, repadding rows to 48-element pitch
// (pad_interval=3 -> pad after every 16 DWORDs, pad_amount=7 -> 8 DWORDs).
// D# layout per CDNA5 ISA. Tracked by TENSORcnt.
static __device__ __forceinline__ void tdm_tile_32x16(unsigned lds, unsigned long long ga,
                                                      unsigned stride) {
    v4u g0 = { 1u,                                       // count=1 (valid descriptor)
               lds,                                      // lds_addr (bytes)
               (unsigned)ga,                             // global_addr[31:0]
               ((unsigned)(ga >> 32) & 0x01FFFFFFu) | 0x80000000u };  // ga[56:32] | type=2
    v8u g1 = { (1u << 16) | (1u << 20) | (3u << 22) | (7u << 25),
               //  data_size=2B | pad_enable | pad_interval=16dw | pad_amount=8dw
               (32u << 16),                              // tensor_dim0 = 32 (elements)
               (16u << 16),                              // tensor_dim1 = 16 (rows)
               (32u << 16),                              // tile_dim0 = 32
               16u,                                      // tile_dim1 = 16
               stride,                                   // tensor_dim0_stride[31:0]
               0u, 0u };
    asm volatile("tensor_load_to_lds %0, %1" :: "s"(g0), "s"(g1) : "memory");
}

// ---------------------------------------------------------------------------
// Weight repack (all B operands stored TRANSPOSED, [N][K] row-major):
// Wq/Wk/Wv [H, C, hs] f32  ->  [H*hs, C] bf16
// ---------------------------------------------------------------------------
__global__ void pack_hcd_t(const float* __restrict__ src, unsigned short* __restrict__ dst, int n) {
    int i = blockIdx.x * blockDim.x + threadIdx.x;
    if (i >= n) return;
    int d = i % HS;
    int c = (i / HS) % CC;
    int h = i / (HS * CC);
    dst[((size_t)h * HS + d) * CC + c] = f2bf(src[i]);
}

// src [K, N] f32 -> dst [N, K] bf16
__global__ void cvt_bf_t(const float* __restrict__ src, unsigned short* __restrict__ dst,
                         int K, int N) {
    int i = blockIdx.x * blockDim.x + threadIdx.x;
    if (i >= K * N) return;
    int kk = i / N, nn = i % N;
    dst[(size_t)nn * K + kk] = f2bf(src[i]);
}

// ---------------------------------------------------------------------------
// LayerNorm over C=1024, one block per row, bf16 output
// ---------------------------------------------------------------------------
__global__ __launch_bounds__(256) void lnorm(const float* __restrict__ x,
                                             const float* __restrict__ g,
                                             const float* __restrict__ be,
                                             unsigned short* __restrict__ xn) {
    __shared__ float red[256];
    const int row = blockIdx.x, tid = threadIdx.x;
    const float* xr = x + (size_t)row * CC;
    float xv[4];
    float s = 0.f;
#pragma unroll
    for (int j = 0; j < 4; ++j) { xv[j] = xr[tid + j * 256]; s += xv[j]; }
    red[tid] = s;
    __syncthreads();
    for (int st = 128; st > 0; st >>= 1) {
        if (tid < st) red[tid] += red[tid + st];
        __syncthreads();
    }
    float mu = red[0] * (1.0f / CC);
    __syncthreads();
    float vs = 0.f;
#pragma unroll
    for (int j = 0; j < 4; ++j) { float d = xv[j] - mu; vs += d * d; }
    red[tid] = vs;
    __syncthreads();
    for (int st = 128; st > 0; st >>= 1) {
        if (tid < st) red[tid] += red[tid + st];
        __syncthreads();
    }
    float rstd = rsqrtf(red[0] * (1.0f / CC) + 1e-5f);
#pragma unroll
    for (int j = 0; j < 4; ++j) {
        int c = tid + j * 256;
        xn[(size_t)row * CC + c] = f2bf((xv[j] - mu) * rstd * g[c] + be[c]);
    }
}

// ---------------------------------------------------------------------------
// bf16 GEMM: out = epi( A[M,K] * Bt[N,K]^T ), 128x128 tile, BK=32, 8 waves.
// Tiles staged by the Tensor Data Mover (each wave DMAs its own 16-row slice
// of the A and B tiles), double buffered on TENSORcnt. 8 WMMAs/wave/k-step.
// out_mode: 0 = f32, 1 = bf16, 2 = bf16 transposed to [b,h,d,t] (V for attention)
// ---------------------------------------------------------------------------
__global__ __launch_bounds__(256) void gemm128(const unsigned short* __restrict__ A,
                                               const unsigned short* __restrict__ Bt,
                                               const float* __restrict__ bias,
                                               const float* __restrict__ resid,
                                               void* __restrict__ outp,
                                               int M, int N, int K,
                                               int out_mode, int relu) {
    __shared__ __align__(16) unsigned short As[2][128 * 48];  // [m][k], 48-elem pitch
    __shared__ __align__(16) unsigned short Bs[2][128 * 48];  // [n][k]
    const int tid = threadIdx.x;
    const int lane = tid & 31, half = lane >> 4, ln = lane & 15;
    const int wv = __builtin_amdgcn_readfirstlane(tid) >> 5;   // scalar wave id
    const int mrow = (wv & 3) * 32, ncol = (wv >> 2) * 64;
    const int m0 = blockIdx.y * 128, n0 = blockIdx.x * 128;

    v8f acc[2][4];
#pragma unroll
    for (int mi = 0; mi < 2; ++mi)
#pragma unroll
        for (int ni = 0; ni < 4; ++ni)
#pragma unroll
            for (int i = 0; i < 8; ++i) acc[mi][ni][i] = 0.f;

    // each wave DMAs 16 rows of A and 16 rows of Bt for this k-step
    auto stage = [&](int buf, int k0) {
        unsigned lds_a = (unsigned)(size_t)(&As[buf][0]) + (unsigned)(wv * 16 * 96);
        tdm_tile_32x16(lds_a,
                       (unsigned long long)(size_t)(A + (size_t)(m0 + wv * 16) * K + k0),
                       (unsigned)K);
        unsigned lds_b = (unsigned)(size_t)(&Bs[buf][0]) + (unsigned)(wv * 16 * 96);
        tdm_tile_32x16(lds_b,
                       (unsigned long long)(size_t)(Bt + (size_t)(n0 + wv * 16) * K + k0),
                       (unsigned)K);
    };

    const int steps = K >> 5;
    stage(0, 0);
    for (int kt = 0; kt < steps; ++kt) {
        const int buf = kt & 1;
        const bool more = (kt + 1 < steps);
        if (more) {
            stage(buf ^ 1, (kt + 1) << 5);
            __builtin_amdgcn_s_wait_tensorcnt(2);  // previous stage (2 TDM ops/wave) retired
        } else {
            __builtin_amdgcn_s_wait_tensorcnt(0);
        }
        __syncthreads();

        v16bf af[2];
#pragma unroll
        for (int mi = 0; mi < 2; ++mi)
            af[mi] = ldfrag<48>(As[buf], mrow + 16 * mi + ln, 8 * half);
#pragma unroll
        for (int ni = 0; ni < 4; ++ni) {
            v16bf bf = ldfrag<48>(Bs[buf], ncol + 16 * ni + ln, 8 * half);
#pragma unroll
            for (int mi = 0; mi < 2; ++mi)
                acc[mi][ni] = wmma_bf16(af[mi], bf, acc[mi][ni]);
        }
        __syncthreads();
    }

#pragma unroll
    for (int mi = 0; mi < 2; ++mi)
#pragma unroll
        for (int ni = 0; ni < 4; ++ni)
#pragma unroll
            for (int i = 0; i < 8; ++i) {
                int m = m0 + mrow + 16 * mi + i + 8 * half;
                int n = n0 + ncol + 16 * ni + ln;
                float val = acc[mi][ni][i];
                if (bias)  val += bias[n];
                if (resid) val += resid[(size_t)m * N + n];
                if (relu)  val = fmaxf(val, 0.f);
                if (out_mode == 0) {
                    ((float*)outp)[(size_t)m * N + n] = val;
                } else if (out_mode == 1) {
                    ((unsigned short*)outp)[(size_t)m * N + n] = f2bf(val);
                } else {  // V: [b, t] x [h, d] -> [b, h, d, t]
                    int bq = m / TT, t = m % TT, hh = n / HS, d = n % HS;
                    ((unsigned short*)outp)[(((size_t)bq * HH + hh) * HS + d) * TT + t] = f2bf(val);
                }
            }
}

// ---------------------------------------------------------------------------
// Causal flash attention, one block = (b, h, 64 query rows), 4 waves.
// q,k bf16 [B,T,H,hs] (row stride C); vT bf16 [B,H,hs,T]; o bf16 [B,T,H,hs].
// K/V tiles double-buffered through async global->LDS (ASYNCcnt).
// ---------------------------------------------------------------------------
__global__ __launch_bounds__(128) void attn64(const unsigned short* __restrict__ q,
                                              const unsigned short* __restrict__ k,
                                              const unsigned short* __restrict__ vT,
                                              unsigned short* __restrict__ o) {
    __shared__ __align__(16) unsigned short kt[2][64 * 72];  // K tile [s][d]
    __shared__ __align__(16) unsigned short vt[2][64 * 72];  // V tile [d][s]
    __shared__ __align__(16) unsigned short pt[64 * 72];     // P [m][s]
    const int tid = threadIdx.x;
    const int wave = tid >> 5, lane = tid & 31, half = lane >> 4, ln = lane & 15;
    const int b = blockIdx.z, h = blockIdx.y, it = blockIdx.x;
    const int t0 = it * 64;
    const float scale = 0.125f;  // hs^-0.5

    // Q operand fragments straight from global (16B-aligned runs)
    v16bf aq[2];
    {
        int m = t0 + 16 * wave + ln;
        size_t base = ((size_t)(b * TT + m)) * CC + h * HS;
#pragma unroll
        for (int ks = 0; ks < 2; ++ks) {
            AF f;
            f.q[0] = *(const uint4*)(q + base + ks * 32 + 8 * half);
            f.q[1] = *(const uint4*)(q + base + ks * 32 + 16 + 8 * half);
            aq[ks] = f.v;
        }
    }

    auto stage = [&](int buf, int jt) {
#pragma unroll
        for (int c = 0; c < 4; ++c) {
            int idx = tid * 32 + c * 8;
            int r = idx >> 6, cc = idx & 63;
            // K: rows s contiguous over d
            async_b128(kt[buf] + r * 72 + cc,
                       k + ((size_t)(b * TT + jt * 64 + r)) * CC + h * HS + cc);
            // V^T: rows d contiguous over s
            async_b128(vt[buf] + r * 72 + cc,
                       vT + (((size_t)(b * HH + h)) * HS + r) * TT + jt * 64 + cc);
        }
    };

    float rm[8], rl[8];
    v8f oacc[4];
#pragma unroll
    for (int i = 0; i < 8; ++i) { rm[i] = -1e30f; rl[i] = 0.f; }
#pragma unroll
    for (int nt = 0; nt < 4; ++nt)
#pragma unroll
        for (int i = 0; i < 8; ++i) oacc[nt][i] = 0.f;

    stage(0, 0);
    for (int jt = 0; jt <= it; ++jt) {
        const int buf = jt & 1;
        const bool more = (jt < it);
        if (more) stage(buf ^ 1, jt + 1);
        wait_async(more ? 8 : 0);
        __syncthreads();

        // S = Q * K^T
        v8f sa[4];
#pragma unroll
        for (int nt = 0; nt < 4; ++nt)
#pragma unroll
            for (int i = 0; i < 8; ++i) sa[nt][i] = 0.f;
#pragma unroll
        for (int ks = 0; ks < 2; ++ks)
#pragma unroll
            for (int nt = 0; nt < 4; ++nt) {
                v16bf bf = ldfrag<72>(kt[buf], 16 * nt + ln, ks * 32 + 8 * half);
                sa[nt] = wmma_bf16(aq[ks], bf, sa[nt]);
            }

        // online softmax (D-layout: m = i + 8*half within wave tile, n = ln + 16*nt)
#pragma unroll
        for (int i = 0; i < 8; ++i) {
            int mrow = t0 + 16 * wave + i + 8 * half;
            float tmax = -1e30f;
#pragma unroll
            for (int nt = 0; nt < 4; ++nt) {
                float sv = sa[nt][i] * scale;
                if (jt == it && (jt * 64 + 16 * nt + ln) > mrow) sv = -1e30f;
                sa[nt][i] = sv;
                tmax = fmaxf(tmax, sv);
            }
#pragma unroll
            for (int off = 1; off < 16; off <<= 1)
                tmax = fmaxf(tmax, __shfl_xor(tmax, off));
            float mn = fmaxf(rm[i], tmax);
            float al = __expf(rm[i] - mn);
            float tsum = 0.f;
#pragma unroll
            for (int nt = 0; nt < 4; ++nt) {
                float p = __expf(sa[nt][i] - mn);
                tsum += p;
                pt[(16 * wave + i + 8 * half) * 72 + 16 * nt + ln] = f2bf(p);
            }
#pragma unroll
            for (int off = 1; off < 16; off <<= 1)
                tsum += __shfl_xor(tsum, off);
            rl[i] = rl[i] * al + tsum;
            rm[i] = mn;
#pragma unroll
            for (int nt = 0; nt < 4; ++nt) oacc[nt][i] *= al;
        }
        __syncthreads();

        // O += P * V
#pragma unroll
        for (int ks = 0; ks < 2; ++ks) {
            v16bf ap = ldfrag<72>(pt, 16 * wave + ln, ks * 32 + 8 * half);
#pragma unroll
            for (int nt = 0; nt < 4; ++nt) {
                v16bf bv = ldfrag<72>(vt[buf], 16 * nt + ln, ks * 32 + 8 * half);
                oacc[nt] = wmma_bf16(ap, bv, oacc[nt]);
            }
        }
        __syncthreads();
    }

    // normalize and write bf16 output in [B,T,H,hs]
#pragma unroll
    for (int i = 0; i < 8; ++i) {
        float inv = 1.0f / rl[i];
        int m = t0 + 16 * wave + i + 8 * half;
        size_t base = ((size_t)(b * TT + m)) * CC + h * HS;
#pragma unroll
        for (int nt = 0; nt < 4; ++nt)
            o[base + 16 * nt + ln] = f2bf(oacc[nt][i] * inv);
    }
}

// ---------------------------------------------------------------------------
// Host-side orchestration
// ---------------------------------------------------------------------------
extern "C" void kernel_launch(void* const* d_in, const int* in_sizes, int n_in,
                              void* d_out, int out_size, void* d_ws, size_t ws_size,
                              hipStream_t stream) {
    const float* x   = (const float*)d_in[0];
    const float* Wq  = (const float*)d_in[1];
    const float* Wk  = (const float*)d_in[2];
    const float* Wv  = (const float*)d_in[3];
    const float* Wo  = (const float*)d_in[4];
    const float* bo  = (const float*)d_in[5];
    const float* g1  = (const float*)d_in[6];
    const float* be1 = (const float*)d_in[7];
    const float* g2  = (const float*)d_in[8];
    const float* be2 = (const float*)d_in[9];
    const float* W1  = (const float*)d_in[10];
    const float* b1  = (const float*)d_in[11];
    const float* W2  = (const float*)d_in[12];
    const float* b2  = (const float*)d_in[13];

    char* ws = (char*)d_ws;
    size_t off = 0;
    auto take = [&](size_t bytes) {
        void* p = ws + off;
        off += (bytes + 255) & ~(size_t)255;
        return p;
    };
    unsigned short* Wq_b = (unsigned short*)take((size_t)CC * CC * 2);      // [N=C][K=C]
    unsigned short* Wk_b = (unsigned short*)take((size_t)CC * CC * 2);
    unsigned short* Wv_b = (unsigned short*)take((size_t)CC * CC * 2);
    unsigned short* Wo_b = (unsigned short*)take((size_t)CC * CC * 2);      // [N=C][K=C]
    unsigned short* W1_b = (unsigned short*)take((size_t)CC * FFDIM * 2);   // [N=4C][K=C]
    unsigned short* W2_b = (unsigned short*)take((size_t)FFDIM * CC * 2);   // [N=C][K=4C]
    unsigned short* xn1  = (unsigned short*)take((size_t)MMROWS * CC * 2);
    unsigned short* qb   = (unsigned short*)take((size_t)MMROWS * CC * 2);
    unsigned short* kb   = (unsigned short*)take((size_t)MMROWS * CC * 2);
    unsigned short* vTb  = (unsigned short*)take((size_t)MMROWS * CC * 2);  // [B,H,hs,T]
    unsigned short* attn = (unsigned short*)take((size_t)MMROWS * CC * 2);
    float*          x1   = (float*)take((size_t)MMROWS * CC * 4);
    unsigned short* xn2  = (unsigned short*)take((size_t)MMROWS * CC * 2);
    unsigned short* hbuf = (unsigned short*)take((size_t)MMROWS * FFDIM * 2);

    const int nW  = HH * CC * HS;          // 1M
    const int nW1 = CC * FFDIM;            // 4M
    pack_hcd_t<<<(nW + 255) / 256, 256, 0, stream>>>(Wq, Wq_b, nW);
    pack_hcd_t<<<(nW + 255) / 256, 256, 0, stream>>>(Wk, Wk_b, nW);
    pack_hcd_t<<<(nW + 255) / 256, 256, 0, stream>>>(Wv, Wv_b, nW);
    cvt_bf_t<<<(CC * CC + 255) / 256, 256, 0, stream>>>(Wo, Wo_b, CC, CC);
    cvt_bf_t<<<(nW1 + 255) / 256, 256, 0, stream>>>(W1, W1_b, CC, FFDIM);
    cvt_bf_t<<<(nW1 + 255) / 256, 256, 0, stream>>>(W2, W2_b, FFDIM, CC);

    // ln1
    lnorm<<<MMROWS, 256, 0, stream>>>(x, g1, be1, xn1);

    // Q, K, V projections
    dim3 gQ(CC / 128, MMROWS / 128);
    gemm128<<<gQ, 256, 0, stream>>>(xn1, Wq_b, nullptr, nullptr, qb,  MMROWS, CC, CC, 1, 0);
    gemm128<<<gQ, 256, 0, stream>>>(xn1, Wk_b, nullptr, nullptr, kb,  MMROWS, CC, CC, 1, 0);
    gemm128<<<gQ, 256, 0, stream>>>(xn1, Wv_b, nullptr, nullptr, vTb, MMROWS, CC, CC, 2, 0);

    // causal flash attention
    dim3 gA(TT / 64, HH, BB);
    attn64<<<gA, 128, 0, stream>>>(qb, kb, vTb, attn);

    // output projection + bias + residual -> x1 (f32)
    gemm128<<<gQ, 256, 0, stream>>>(attn, Wo_b, bo, x, x1, MMROWS, CC, CC, 0, 0);

    // ln2
    lnorm<<<MMROWS, 256, 0, stream>>>(x1, g2, be2, xn2);

    // FFN up: relu(xn2 @ W1 + b1) -> bf16
    dim3 gU(FFDIM / 128, MMROWS / 128);
    gemm128<<<gU, 256, 0, stream>>>(xn2, W1_b, b1, nullptr, hbuf, MMROWS, FFDIM, CC, 1, 1);

    // FFN down + bias + residual -> d_out (f32)
    gemm128<<<gQ, 256, 0, stream>>>(hbuf, W2_b, b2, x1, d_out, MMROWS, CC, FFDIM, 0, 0);
}
